// Actor_84344567759689
// MI455X (gfx1250) — matrix operand order repaired and checked by reading, not compiled, not run on hardware
//
#include <hip/hip_runtime.h>
#include <hip/hip_bf16.h>
#include <math.h>

typedef __bf16 bf16;
typedef __attribute__((ext_vector_type(16))) __bf16 v16bf;
typedef __attribute__((ext_vector_type(8)))  float  v8f;

#define B_  32
#define L_  2048
#define E_  256
#define H_  128
#define C_  128
#define NF  768            // 6*H
#define MROWS (B_*L_)      // 65536

__device__ __forceinline__ float bf2f(bf16 x){ return (float)x; }
__device__ __forceinline__ bf16  f2bf(float x){ return (bf16)x; }
__device__ __forceinline__ float sigm(float x){ return 1.0f/(1.0f + __expf(-x)); }

// CDNA5 16-bit A-matrix 16x32 layout: lanes 0-15 hold K={2i,2i+1} (i<4) /
// {16+2(i-4),...} ; lanes 16-31 shifted by +8 within each group.
__device__ __forceinline__ int a_krel(int i, int hi){ return 2*(i&3) + 16*(i>>2) + 8*hi; }
// CDNA5 16-bit B-matrix 32x16 layout: same K order as A (lanes 16-31 shifted by +8).
__device__ __forceinline__ int b_krel(int i, int hi){ return 2*(i&3) + 16*(i>>2) + 8*hi; }

__device__ __forceinline__ v8f wmma_bf16(v16bf a, v16bf b, v8f c){
  return __builtin_amdgcn_wmma_f32_16x16x32_bf16(false, a, false, b, (short)0, c, false, false);
}

// Async global->LDS 16B copy (CDNA5, tracked with ASYNCcnt). The LDS address
// is the low 32 bits of the generic pointer (LDS aperture truncation rule).
__device__ __forceinline__ void async_cp16(void* ldsPtr, const void* gPtr){
  unsigned lds = (unsigned)(size_t)ldsPtr;
  asm volatile("global_load_async_to_lds_b128 %0, %1, off"
               :: "v"(lds), "v"(gPtr) : "memory");
}
__device__ __forceinline__ void wait_async0(){
  asm volatile("s_wait_asynccnt 0x0" ::: "memory");
}

// ---------------------------------------------------------------------------
// Kernel 0: convert scan weights fp32 -> bf16 (row-major, K contiguous)
// ---------------------------------------------------------------------------
__global__ __launch_bounds__(256) void cvt_weights_kernel(
    const float* __restrict__ Wp1, const float* __restrict__ Wih,
    const float* __restrict__ Whh,
    bf16* __restrict__ wp1bf, bf16* __restrict__ wihbf, bf16* __restrict__ whhbf)
{
  int i = blockIdx.x * 256 + threadIdx.x;
  if (i < 128*256)  wp1bf[i] = f2bf(Wp1[i]);   // [128][256]
  if (i < 512*256)  wihbf[i] = f2bf(Wih[i]);   // [512][256]
  if (i < 512*128)  whhbf[i] = f2bf(Whh[i]);   // [512][128]
}

// ---------------------------------------------------------------------------
// Kernel 1: cf = relu(memory[:,1:-1,:] @ W1^T + b1)  -> bf16 [B*L, 768]
// One wave per 16-row M tile x 4 N tiles. K=256 in 8 WMMA steps.
// ---------------------------------------------------------------------------
__global__ __launch_bounds__(256) void cf_gemm_kernel(
    const float* __restrict__ mem, const float* __restrict__ W1,
    const float* __restrict__ b1, bf16* __restrict__ cfW)
{
  const int lane   = threadIdx.x & 31;
  const int hi     = lane >> 4;
  const int lanelo = lane & 15;
  const int waveId = blockIdx.x * 8 + (threadIdx.x >> 5);
  const int mtile  = waveId / 12;       // 0..4095
  const int ngrp   = waveId % 12;       // 0..11 (4 N tiles each)

  const int m  = mtile*16 + lanelo;     // A-fragment row
  const int bb = m >> 11;
  const int ll = m & 2047;
  const float* __restrict__ arow = mem + (size_t)(bb*(L_+2) + ll + 1) * E_;

  const v8f vz = {0.f,0.f,0.f,0.f,0.f,0.f,0.f,0.f};
  v8f acc[4];
  #pragma unroll
  for (int tn = 0; tn < 4; ++tn) acc[tn] = vz;

  for (int kk = 0; kk < E_; kk += 32) {
    v16bf a;
    #pragma unroll
    for (int i = 0; i < 8; ++i) {
      int k = kk + a_krel(i, hi);
      a[2*i]   = f2bf(arow[k]);
      a[2*i+1] = f2bf(arow[k+1]);
    }
    #pragma unroll
    for (int tn = 0; tn < 4; ++tn) {
      int n = (ngrp*4 + tn)*16 + lanelo;
      const float* __restrict__ brow = W1 + (size_t)n * E_;   // B[k,n] = W1[n,k]
      v16bf bm;
      #pragma unroll
      for (int i = 0; i < 8; ++i) {
        int k = kk + b_krel(i, hi);
        bm[2*i]   = f2bf(brow[k]);
        bm[2*i+1] = f2bf(brow[k+1]);
      }
      acc[tn] = wmma_bf16(a, bm, acc[tn]);
    }
  }

  // epilogue: +b1, relu, store bf16. C layout: M = i + 8*hi, N = lanelo.
  #pragma unroll
  for (int tn = 0; tn < 4; ++tn) {
    int n = (ngrp*4 + tn)*16 + lanelo;
    float bias = b1[n];
    #pragma unroll
    for (int i = 0; i < 8; ++i) {
      int mrow = mtile*16 + i + 8*hi;
      float v = acc[tn][i] + bias;
      v = v > 0.f ? v : 0.f;
      cfW[(size_t)mrow*NF + n] = f2bf(v);
    }
  }
}

// ---------------------------------------------------------------------------
// Kernel 2: sequential scan, single workgroup (8 wave32), t = 1..L-1
// cf_t staged in LDS via double-buffered async copies (ASYNCcnt).
// ---------------------------------------------------------------------------
__global__ __launch_bounds__(256) void scan_kernel(
    const bf16* __restrict__ cfW, const unsigned char* __restrict__ mask,
    const int* __restrict__ length,
    const float* __restrict__ bp1, const float* __restrict__ Wp2,
    const float* __restrict__ bp2, const float* __restrict__ bih,
    const float* __restrict__ bhh,
    const bf16* __restrict__ wp1bf, const bf16* __restrict__ wihbf,
    const bf16* __restrict__ whhbf,
    float* __restrict__ out)
{
  __shared__ bf16  sH[B_*C_];        // h_t, bf16           (8 KB)
  __shared__ bf16  sChosen[B_*256];  // chosen x rows       (16 KB)
  __shared__ bf16  sGates[B_*512];   // gate preacts        (32 KB)
  __shared__ float sSc[192];         // per (b,choice) score partials
  __shared__ int   sIdx[B_];
  extern __shared__ char dynLds[];   // 2 x 48KB cf_t double buffer

  const int tid    = threadIdx.x;
  const int lane   = tid & 31;
  const int wave   = tid >> 5;
  const int hi     = lane >> 4;
  const int lanelo = lane & 15;
  const v8f vz = {0.f,0.f,0.f,0.f,0.f,0.f,0.f,0.f};

  float creg[16];                    // cell state: thread tid owns units tid*16..+15
  #pragma unroll
  for (int q = 0; q < 16; ++q) creg[q] = 0.f;
  float lpAcc = 0.f;
  int   lenb  = (tid < B_) ? length[tid] : 0;

  for (int z = tid; z < B_*C_; z += 256) sH[z] = f2bf(0.f);
  if (tid < 192) sSc[tid] = 0.f;

  // initial prefetch of cf_{t=1} into buffer 1 (192 rows x 256B = 3072 x 16B)
  {
    bf16* dst0 = (bf16*)dynLds + (size_t)(192*H_);   // buffer 1
    #pragma unroll
    for (int it = 0; it < 12; ++it) {
      int z  = tid + 256*it;
      int r  = z >> 4;                 // row (= b*6 + j)
      int cb = (z & 15) << 4;          // byte within row
      int bA = r / 6, jA = r % 6;
      const char* g = (const char*)(cfW + (size_t)((bA*L_ + 1)*6 + jA)*H_) + cb;
      async_cp16((char*)dst0 + r*256 + cb, g);
    }
  }
  wait_async0();
  __syncthreads();

  for (int t = 1; t < L_; ++t) {
    const bf16* cfP = (const bf16*)dynLds + (size_t)(t & 1) * (192*H_);

    // issue async prefetch of cf_{t+1} into the other buffer (overlaps compute)
    if (t + 1 < L_) {
      bf16* dst = (bf16*)dynLds + (size_t)((t + 1) & 1) * (192*H_);
      #pragma unroll
      for (int it = 0; it < 12; ++it) {
        int z  = tid + 256*it;
        int r  = z >> 4;
        int cb = (z & 15) << 4;
        int bA = r / 6, jA = r % 6;
        const char* g = (const char*)(cfW + (size_t)((bA*L_ + t + 1)*6 + jA)*H_) + cb;
        async_cp16((char*)dst + r*256 + cb, g);
      }
    }

    // ---- Phase A: hid = relu(x @ Wp1^T + bp1); fused sc partial reduction.
    // wave -> fixed ntile; Wp1 B-fragments hoisted (depend only on kk).
    {
      const int nA   = wave*16 + lanelo;
      const float wp2n = Wp2[nA];
      const float bp1n = bp1[nA];
      const bf16* __restrict__ wrow = wp1bf + (size_t)nA * 256;
      v16bf bfr[8];
      #pragma unroll
      for (int kk = 0; kk < 8; ++kk) {
        #pragma unroll
        for (int i = 0; i < 8; ++i) {
          int k = kk*32 + b_krel(i, hi);
          bfr[kk][2*i] = wrow[k]; bfr[kk][2*i+1] = wrow[k+1];
        }
      }
      for (int mt = 0; mt < 12; ++mt) {
        const int mA = mt*16 + lanelo;            // x row = b*6 + choice
        const int bA = mA / 6;
        v8f acc = vz;
        #pragma unroll
        for (int kk = 0; kk < 8; ++kk) {
          v16bf a;
          #pragma unroll
          for (int i = 0; i < 8; ++i) {
            int k = kk*32 + a_krel(i, hi);
            if (k < 128) { a[2*i] = cfP[mA*H_ + k];      a[2*i+1] = cfP[mA*H_ + k + 1]; }
            else         { a[2*i] = sH[bA*C_ + (k-128)]; a[2*i+1] = sH[bA*C_ + (k-127)]; }
          }
          acc = wmma_bf16(a, bfr[kk], acc);
        }
        // epilogue: relu(hid)*Wp2[n], reduce 16 lanes sharing M, atomic into sSc
        #pragma unroll
        for (int i = 0; i < 8; ++i) {
          float v = acc[i] + bp1n;
          v = v > 0.f ? v : 0.f;
          float s = v * wp2n;
          s += __shfl_xor(s, 1);
          s += __shfl_xor(s, 2);
          s += __shfl_xor(s, 4);
          s += __shfl_xor(s, 8);
          if (lanelo == 0) atomicAdd(&sSc[mt*16 + i + 8*hi], s);
        }
      }
    }
    __syncthreads();

    // ---- Phase C1: sampling (one thread per batch row)
    if (tid < B_) {
      const int b = tid;
      float logits[6], ex[6];
      float mx = -1e30f;
      #pragma unroll
      for (int j = 0; j < 6; ++j) {
        float sc = sSc[b*6 + j] + bp2[0];
        sc = sc > 0.f ? sc : 0.f;
        bool mk = mask[(size_t)(b*L_ + t)*6 + j] != 0;
        logits[j] = mk ? sc : -1e9f;
        mx = fmaxf(mx, logits[j]);
      }
      float sum = 0.f;
      #pragma unroll
      for (int j = 0; j < 6; ++j) { ex[j] = __expf(logits[j] - mx); sum += ex[j]; }
      unsigned s = (unsigned)t * 0x9E3779B9u ^ ((unsigned)b * 0x85EBCA6Bu + 0xC2B2AE35u);
      s ^= s >> 16; s *= 0x7feb352du; s ^= s >> 15; s *= 0x846ca68bu; s ^= s >> 16;
      float u = (float)(s >> 8) * (1.0f/16777216.0f);
      float target = u * sum, cum = 0.f;
      int idx = 5; bool done = false;
      #pragma unroll
      for (int j = 0; j < 6; ++j) {
        cum += ex[j];
        if (!done && cum >= target) { idx = j; done = true; }
      }
      float logp = (logits[idx] - mx) - __logf(sum);
      if (lenb > t) lpAcc += logp;
      sIdx[b] = idx;
      out[B_ + (t-1)*B_ + b] = (float)idx;
    }
    __syncthreads();

    // ---- Phase C2: gather chosen = [cf(b,t,idx,:), h(b,:)]  (LDS -> LDS)
    for (int z = tid; z < B_*256; z += 256) {
      int b = z >> 8, k = z & 255;
      int idx = sIdx[b];
      sChosen[z] = (k < 128) ? cfP[(b*6 + idx)*H_ + k] : sH[b*C_ + (k-128)];
    }
    __syncthreads();

    // ---- Phase D: gates = chosen @ Wih^T + h @ Whh^T + bih + bhh (M=32,N=512)
    for (int j = 0; j < 8; ++j) {
      const int tileId = wave + 8*j;        // 64 tiles total
      const int mtile  = tileId >> 5;       // 0..1
      const int ntile  = tileId & 31;       // 0..31
      const int mD = mtile*16 + lanelo;     // < 32 (== batch row)
      const int nD = ntile*16 + lanelo;
      v8f acc = vz;
      const bf16* __restrict__ wrowI = wihbf + (size_t)nD * 256;
      #pragma unroll
      for (int kk = 0; kk < 8; ++kk) {
        v16bf a, bm;
        #pragma unroll
        for (int i = 0; i < 8; ++i) {
          int k = kk*32 + a_krel(i, hi);
          a[2*i] = sChosen[mD*256 + k]; a[2*i+1] = sChosen[mD*256 + k + 1];
        }
        #pragma unroll
        for (int i = 0; i < 8; ++i) {
          int k = kk*32 + b_krel(i, hi);
          bm[2*i] = wrowI[k]; bm[2*i+1] = wrowI[k+1];
        }
        acc = wmma_bf16(a, bm, acc);
      }
      const bf16* __restrict__ wrowH = whhbf + (size_t)nD * 128;
      #pragma unroll
      for (int kk = 0; kk < 4; ++kk) {
        v16bf a, bm;
        #pragma unroll
        for (int i = 0; i < 8; ++i) {
          int k = kk*32 + a_krel(i, hi);
          a[2*i] = sH[mD*C_ + k]; a[2*i+1] = sH[mD*C_ + k + 1];
        }
        #pragma unroll
        for (int i = 0; i < 8; ++i) {
          int k = kk*32 + b_krel(i, hi);
          bm[2*i] = wrowH[k]; bm[2*i+1] = wrowH[k+1];
        }
        acc = wmma_bf16(a, bm, acc);
      }
      float bias = bih[nD] + bhh[nD];
      #pragma unroll
      for (int i = 0; i < 8; ++i) {
        int mr = mtile*16 + i + 8*hi;       // < 32
        sGates[mr*512 + nD] = f2bf(acc[i] + bias);
      }
    }
    __syncthreads();

    // ---- Phase E: LSTM elementwise; write h_{t+1}; re-zero sSc
    #pragma unroll
    for (int q = 0; q < 16; ++q) {
      int z = tid*16 + q;
      int b = z >> 7, u2 = z & 127;
      float ig = bf2f(sGates[b*512 + u2]);
      float fg = bf2f(sGates[b*512 + 128 + u2]);
      float gg = bf2f(sGates[b*512 + 256 + u2]);
      float og = bf2f(sGates[b*512 + 384 + u2]);
      float c2 = sigm(fg)*creg[q] + sigm(ig)*tanhf(gg);
      float h2 = sigm(og)*tanhf(c2);
      creg[q] = c2;
      sH[b*C_ + u2] = f2bf(h2);
    }
    if (tid < 192) sSc[tid] = 0.f;

    // drain this wave's async prefetch before anyone reads the new buffer
    wait_async0();
    __syncthreads();
  }

  if (tid < B_) out[tid] = lpAcc;
}

// ---------------------------------------------------------------------------
extern "C" void kernel_launch(void* const* d_in, const int* in_sizes, int n_in,
                              void* d_out, int out_size, void* d_ws, size_t ws_size,
                              hipStream_t stream) {
  (void)in_sizes; (void)n_in; (void)out_size; (void)ws_size;
  const float*         mem    = (const float*)d_in[0];
  const unsigned char* mask   = (const unsigned char*)d_in[1];
  const int*           length = (const int*)d_in[2];
  const float*         W1     = (const float*)d_in[3];
  const float*         b1     = (const float*)d_in[4];
  const float*         Wp1    = (const float*)d_in[5];
  const float*         bp1    = (const float*)d_in[6];
  const float*         Wp2    = (const float*)d_in[7];
  const float*         bp2    = (const float*)d_in[8];
  const float*         Wih    = (const float*)d_in[9];
  const float*         Whh    = (const float*)d_in[10];
  const float*         bih    = (const float*)d_in[11];
  const float*         bhh    = (const float*)d_in[12];
  float* out = (float*)d_out;

  // workspace layout: cf bf16 [65536*768] then bf16 weights
  bf16* cfW   = (bf16*)d_ws;
  size_t cfE  = (size_t)MROWS * NF;
  bf16* wp1bf = cfW + cfE;
  bf16* wihbf = wp1bf + 128*256;
  bf16* whhbf = wihbf + 512*256;

  cvt_weights_kernel<<<512, 256, 0, stream>>>(Wp1, Wih, Whh, wp1bf, wihbf, whhbf);
  // 4096 M-tiles * 12 N-groups = 49152 waves / 8 waves per block
  cf_gemm_kernel<<<6144, 256, 0, stream>>>(mem, W1, b1, cfW);
  scan_kernel<<<1, 256, 2*192*H_*sizeof(bf16), stream>>>(
      cfW, mask, length, bp1, Wp2, bp2, bih, bhh,
      wp1bf, wihbf, whhbf, out);
}
